// InstantNGP_80530636800214
// MI455X (gfx1250) — matrix operand order repaired and checked by reading, not compiled, not run on hardware
//
#include <hip/hip_runtime.h>
#include <hip/hip_bf16.h>

typedef __attribute__((ext_vector_type(16))) _Float16 v16h;
typedef __attribute__((ext_vector_type(8)))  _Float16 v8h;
typedef __attribute__((ext_vector_type(8)))  float    v8f;

#define NW    8      // waves per block (wave32)
#define BLOCK 256

// ---- level metadata matching reference _level_meta() ----
// growth = 16^(1/15); res_l = floor(32*growth^l); dense while (res+1)^3 <= 256^3
static constexpr int kResLo[8] = {32, 38, 46, 55, 67, 80, 97, 116};
static constexpr int kResHi[8] = {140, 168, 203, 244, 294, 353, 425, 512};
static constexpr int kOffLo[8] = {0, 35937, 95256, 199079, 374695, 689127, 1220568, 2161760};
static constexpr int kOffHi[8] = {3763373, 6566594, 11393403, 19883067,
                                  34589192, 35113480, 35637768, 36162056};

__device__ __forceinline__ void wsync() {
  // wave-local: make prior LDS stores visible to subsequent LDS loads
  __builtin_amdgcn_wave_barrier();
  asm volatile("s_wait_dscnt 0" ::: "memory");
}

__device__ __forceinline__ v16h cat16(v8h lo, v8h hi) {
  return __builtin_shufflevector(lo, hi, 0,1,2,3,4,5,6,7,8,9,10,11,12,13,14,15);
}

// B fragment (16-bit, KROWSx16 tile, column ncol): lanes 0-15 hold K=koff..koff+15,
// lanes 16-31 hold K=koff+16..koff+31 of column ncol. Weights stored transposed
// wt[n*KROWS + k] so this is two contiguous ds_load_b128.
__device__ __forceinline__ v16h load_b_frag(const _Float16* wt, int krows, int ncol,
                                            int koff, int lane) {
  const _Float16* p = wt + ncol * krows + koff + ((lane < 16) ? 0 : 16);
  v8h lo = *(const v8h*)p;
  v8h hi = *(const v8h*)(p + 8);
  return cat16(lo, hi);
}

// A fragment (16-bit, 16x32 tile starting at column koff of a 16x64 row-major LDS
// activation buffer): lane L row M=L%16; element i -> K = base + i,
// base = koff + (lane<16 ? 0 : 16). Two contiguous ds_load_b128 per lane.
__device__ __forceinline__ v16h load_a_frag(const _Float16* act, int koff, int lane) {
  const _Float16* row = act + (lane & 15) * 64 + koff + ((lane < 16) ? 0 : 16);
  v8h lo = *(const v8h*)row;
  v8h hi = *(const v8h*)(row + 8);
  return cat16(lo, hi);
}

__device__ __forceinline__ v8f wmma16(v16h a, v16h b, v8f c) {
  return __builtin_amdgcn_wmma_f32_16x16x32_f16(false, a, false, b, (short)0, c,
                                                false, false);
}

// 64-wide dense layer: dst[16x64] = act(src[16xKROWS] @ W[KROWSx64] + bias)
// C starts as inline 0; bias folded into the epilogue add (keeps ReLU to one v_max).
template <int KROWS, bool RELU>
__device__ __forceinline__ void layer64(const _Float16* src, const _Float16* wt,
                                        const float* bias, _Float16* dst, int lane) {
  v16h a0 = load_a_frag(src, 0, lane);
  v16h a1{};
  if constexpr (KROWS == 64) a1 = load_a_frag(src, 32, lane);
  const int n0 = lane & 15;
  const int mbase = (lane < 16) ? 0 : 8;
#pragma unroll
  for (int nt = 0; nt < 4; ++nt) {
    const int ncol = nt * 16 + n0;
    v8f acc{};  // inline-0 C operand on the first WMMA
    acc = wmma16(a0, load_b_frag(wt, KROWS, ncol, 0, lane), acc);
    if constexpr (KROWS == 64)
      acc = wmma16(a1, load_b_frag(wt, KROWS, ncol, 32, lane), acc);
    const float b = bias[ncol];
#pragma unroll
    for (int v = 0; v < 8; ++v) {
      float x = acc[v] + b;
      if (RELU) x = fmaxf(x, 0.0f);
      dst[(mbase + v) * 64 + ncol] = (_Float16)x;
    }
  }
}

__global__ __launch_bounds__(BLOCK)
void InstantNGP_kernel(const float* __restrict__ coords, const float* __restrict__ dirs,
                       const float* __restrict__ table,
                       const float* __restrict__ fw1, const float* __restrict__ fb1,
                       const float* __restrict__ fw2, const float* __restrict__ fb2,
                       const float* __restrict__ fw3, const float* __restrict__ fb3,
                       const float* __restrict__ rw1, const float* __restrict__ rb1,
                       const float* __restrict__ rw2, const float* __restrict__ rb2,
                       const float* __restrict__ rw3, const float* __restrict__ rb3,
                       const float* __restrict__ rw4, const float* __restrict__ rb4,
                       float* __restrict__ out, int N) {
  // ---- weights staged once per block, f16, transposed [n][k] ----
  __shared__ __align__(16) _Float16 w1t[64 * 32];
  __shared__ __align__(16) _Float16 w2t[64 * 64];
  __shared__ __align__(16) _Float16 w3t[16 * 64];
  __shared__ __align__(16) _Float16 r1t[64 * 32];   // K padded 31->32 with zeros
  __shared__ __align__(16) _Float16 r2t[64 * 64];
  __shared__ __align__(16) _Float16 r3t[64 * 64];
  __shared__ __align__(16) _Float16 r4t[16 * 64];   // N padded 3->16 with zeros
  __shared__ float fb1s[64], fb2s[64], fb3s[16], rb1s[64], rb2s[64], rb3s[64], rb4s[16];
  __shared__ __align__(16) _Float16 stage[NW][16 * 64];  // per-wave activation tile
  __shared__ __align__(16) float outb[NW][16 * 4];       // per-wave rgb+sigma

  const int tid = threadIdx.x;
  for (int i = tid; i < 64 * 32; i += BLOCK) { int n = i >> 5, k = i & 31; w1t[i] = (_Float16)fw1[k * 64 + n]; }
  for (int i = tid; i < 64 * 64; i += BLOCK) { int n = i >> 6, k = i & 63; w2t[i] = (_Float16)fw2[k * 64 + n]; }
  for (int i = tid; i < 16 * 64; i += BLOCK) { int n = i >> 6, k = i & 63; w3t[i] = (_Float16)fw3[k * 16 + n]; }
  for (int i = tid; i < 64 * 32; i += BLOCK) { int n = i >> 5, k = i & 31; r1t[i] = (_Float16)((k < 31) ? rw1[k * 64 + n] : 0.0f); }
  for (int i = tid; i < 64 * 64; i += BLOCK) { int n = i >> 6, k = i & 63; r2t[i] = (_Float16)rw2[k * 64 + n]; }
  for (int i = tid; i < 64 * 64; i += BLOCK) { int n = i >> 6, k = i & 63; r3t[i] = (_Float16)rw3[k * 64 + n]; }
  for (int i = tid; i < 16 * 64; i += BLOCK) { int n = i >> 6, k = i & 63; r4t[i] = (_Float16)((n < 3) ? rw4[k * 3 + n] : 0.0f); }
  if (tid < 64) { fb1s[tid] = fb1[tid]; fb2s[tid] = fb2[tid]; rb1s[tid] = rb1[tid];
                  rb2s[tid] = rb2[tid]; rb3s[tid] = rb3[tid]; }
  if (tid < 16) { fb3s[tid] = fb3[tid]; rb4s[tid] = (tid < 3) ? rb4[tid] : 0.0f; }
  __syncthreads();

  const int lane = tid & 31;
  const int wv = tid >> 5;
  _Float16* st = stage[wv];
  float* ob = outb[wv];
  const int p = lane & 15;               // point within tile (2 lanes per point)
  const int mbase = (lane < 16) ? 0 : 8; // C/D row base for this lane half
  const int ntiles = (N + 15) >> 4;

  for (int tile = (int)blockIdx.x * NW + wv; tile < ntiles;
       tile += (int)gridDim.x * NW) {
    const int gbase = tile << 4;
    int gp = gbase + p;
    if (gp > N - 1) gp = N - 1;

    // ---------- hash-grid encoding: lane handles 8 levels of point p ----------
    const float cx = coords[gp * 3 + 0];
    const float cy = coords[gp * 3 + 1];
    const float cz = coords[gp * 3 + 2];
#pragma unroll
    for (int j = 0; j < 8; ++j) {
      const int lvl = (lane < 16) ? j : j + 8;
      const int res = (lane < 16) ? kResLo[j] : kResHi[j];
      const int off = (lane < 16) ? kOffLo[j] : kOffHi[j];
      const bool dense = (lane < 16) || (j < 4);
      const float xl = cx * (float)res, yl = cy * (float)res, zl = cz * (float)res;
      const int ix = (int)fminf(fmaxf(floorf(xl), 0.0f), (float)(res - 1));
      const int iy = (int)fminf(fmaxf(floorf(yl), 0.0f), (float)(res - 1));
      const int iz = (int)fminf(fmaxf(floorf(zl), 0.0f), (float)(res - 1));
      const float wx = xl - (float)ix, wy = yl - (float)iy, wz = zl - (float)iz;
      const int r1d = res + 1;
      float e0 = 0.0f, e1 = 0.0f;
#pragma unroll
      for (int c = 0; c < 8; ++c) {
        const int ox = (c >> 2) & 1, oy = (c >> 1) & 1, oz = c & 1;
        const unsigned gx = (unsigned)(ix + ox), gy = (unsigned)(iy + oy),
                       gz = (unsigned)(iz + oz);
        int idx;
        if (dense)
          idx = (int)(gx + gy * (unsigned)r1d + gz * (unsigned)(r1d * r1d));
        else
          idx = (int)((gx * 1u ^ gy * 2654435761u ^ gz * 805459861u) & 524287u);
        const float2 tv = ((const float2*)table)[off + idx];  // global_load_b64 gather
        const float cw = (ox ? wx : 1.0f - wx) * (oy ? wy : 1.0f - wy) *
                         (oz ? wz : 1.0f - wz);
        e0 = fmaf(tv.x, cw, e0);
        e1 = fmaf(tv.y, cw, e1);
      }
      st[p * 64 + 2 * lvl]     = (_Float16)e0;
      st[p * 64 + 2 * lvl + 1] = (_Float16)e1;
    }
    wsync();

    // ---------- density MLP ----------
    layer64<32, true>(st, w1t, fb1s, st, lane);   // 32 -> 64, relu
    wsync();
    layer64<64, true>(st, w2t, fb2s, st, lane);   // 64 -> 64, relu
    wsync();
    {  // 64 -> 16 feat (no relu); sigma = exp(feat[:,0])
      v16h a0 = load_a_frag(st, 0, lane);
      v16h a1 = load_a_frag(st, 32, lane);
      const int n0 = lane & 15;
      v8f acc{};
      acc = wmma16(a0, load_b_frag(w3t, 64, n0, 0, lane), acc);
      acc = wmma16(a1, load_b_frag(w3t, 64, n0, 32, lane), acc);
      const float b = fb3s[n0];
#pragma unroll
      for (int v = 0; v < 8; ++v) {
        const float x = acc[v] + b;
        st[(mbase + v) * 64 + n0] = (_Float16)x;            // feat -> cols 0..15
        if (n0 == 0) ob[(mbase + v) * 4 + 3] = __expf(x);   // sigma
      }
    }
    // ---------- direction positional encoding -> cols 16..31 ----------
    if (lane < 16) {
      const float dx = dirs[gp * 3 + 0], dy = dirs[gp * 3 + 1], dz = dirs[gp * 3 + 2];
      _Float16* r = st + p * 64 + 16;
      r[0] = (_Float16)dx;  r[1] = (_Float16)dy;  r[2] = (_Float16)dz;
      r[3] = (_Float16)__sinf(dx); r[4] = (_Float16)__sinf(2.0f * dx);
      r[5] = (_Float16)__sinf(dy); r[6] = (_Float16)__sinf(2.0f * dy);
      r[7] = (_Float16)__sinf(dz); r[8] = (_Float16)__sinf(2.0f * dz);
      r[9]  = (_Float16)__cosf(dx); r[10] = (_Float16)__cosf(2.0f * dx);
      r[11] = (_Float16)__cosf(dy); r[12] = (_Float16)__cosf(2.0f * dy);
      r[13] = (_Float16)__cosf(dz); r[14] = (_Float16)__cosf(2.0f * dz);
      r[15] = (_Float16)0.0f;  // K pad 31 -> 32
    }
    wsync();

    // ---------- rgb MLP ----------
    layer64<32, true>(st, r1t, rb1s, st, lane);   // 32(pad) -> 64, relu
    wsync();
    layer64<64, true>(st, r2t, rb2s, st, lane);   // 64 -> 64, relu
    wsync();
    layer64<64, true>(st, r3t, rb3s, st, lane);   // 64 -> 64, relu
    wsync();
    {  // 64 -> 16 (3 valid), sigmoid
      v16h a0 = load_a_frag(st, 0, lane);
      v16h a1 = load_a_frag(st, 32, lane);
      const int n0 = lane & 15;
      v8f acc{};
      acc = wmma16(a0, load_b_frag(r4t, 64, n0, 0, lane), acc);
      acc = wmma16(a1, load_b_frag(r4t, 64, n0, 32, lane), acc);
      const float b = rb4s[n0];
      if (n0 < 3) {
#pragma unroll
        for (int v = 0; v < 8; ++v)
          ob[(mbase + v) * 4 + n0] = 1.0f / (1.0f + __expf(-(acc[v] + b)));
      }
    }
    wsync();

    // ---------- write [rgb, sigma] as one b128 per point ----------
    if (lane < 16) {
      const int g = gbase + lane;
      if (g < N) ((float4*)out)[g] = ((const float4*)ob)[lane];
    }
    wsync();  // before next tile overwrites st/ob
  }
}

extern "C" void kernel_launch(void* const* d_in, const int* in_sizes, int n_in,
                              void* d_out, int out_size, void* d_ws, size_t ws_size,
                              hipStream_t stream) {
  const float* coords = (const float*)d_in[0];
  const float* dirs   = (const float*)d_in[1];
  const float* table  = (const float*)d_in[2];
  const float* fw1 = (const float*)d_in[3];
  const float* fb1 = (const float*)d_in[4];
  const float* fw2 = (const float*)d_in[5];
  const float* fb2 = (const float*)d_in[6];
  const float* fw3 = (const float*)d_in[7];
  const float* fb3 = (const float*)d_in[8];
  const float* rw1 = (const float*)d_in[9];
  const float* rb1 = (const float*)d_in[10];
  const float* rw2 = (const float*)d_in[11];
  const float* rb2 = (const float*)d_in[12];
  const float* rw3 = (const float*)d_in[13];
  const float* rb3 = (const float*)d_in[14];
  const float* rw4 = (const float*)d_in[15];
  const float* rb4 = (const float*)d_in[16];
  float* out = (float*)d_out;

  const int N = in_sizes[0] / 3;
  const int ntiles = (N + 15) / 16;
  int blocks = (ntiles + NW - 1) / NW;
  if (blocks > 4096) blocks = 4096;

  InstantNGP_kernel<<<dim3(blocks), dim3(BLOCK), 0, stream>>>(
      coords, dirs, table, fw1, fb1, fw2, fb2, fw3, fb3,
      rw1, rb1, rw2, rb2, rw3, rb3, rw4, rb4, out, N);
}
